// StarPlusTransformer_35923106464140
// MI455X (gfx1250) — compile-verified, matrix operand order checked by reading, not run on hardware
//
#include <hip/hip_runtime.h>
#include <hip/hip_bf16.h>
#include <math.h>
#include <stdint.h>

// ---------------- problem constants ----------------
static constexpr int kB  = 8;
static constexpr int kL  = 4096;
static constexpr int kH  = 256;
static constexpr int kNH = 4;
static constexpr int kHD = 64;
static constexpr int kDI = 1024;
static constexpr int kNL = 6;
static constexpr int kC  = kNH * kHD;       // 256
static constexpr long long kM = (long long)kB * kL;  // 32768 tokens
static constexpr float kScale = 0.125f;     // 1/sqrt(64)
static constexpr float kSlope = 0.01f;      // leaky_relu
static constexpr float kEps   = 1e-5f;

typedef __bf16 bf16_t;
typedef __attribute__((ext_vector_type(16))) __bf16 v16bf;
typedef __attribute__((ext_vector_type(8)))  __bf16 v8bf;
typedef __attribute__((ext_vector_type(8)))  float  v8f;

// ---------------- gfx1250 async-to-LDS path (guarded) ----------------
#if __has_builtin(__builtin_amdgcn_global_load_async_to_lds_b128)
#define HAVE_ASYNC_LDS 1
#else
#define HAVE_ASYNC_LDS 0
#endif

#if HAVE_ASYNC_LDS
// Param 0 (from clang diagnostic): generic pointer to int __attribute__((vector_size(16))).
typedef int v4i_t __attribute__((vector_size(16)));
typedef __attribute__((address_space(3))) v4i_t* lds_v4i_p;

__device__ inline v4i_t* to_g128(const void* p) {
  return (v4i_t*)(uintptr_t)p;
}
// LDS flat addresses carry the LDS byte offset in the low 32 bits.
__device__ inline lds_v4i_p to_lds128(void* p) {
  return (lds_v4i_p)(unsigned)(uintptr_t)p;
}
__device__ inline void wait_asynccnt0() {
#if __has_builtin(__builtin_amdgcn_s_wait_asynccnt)
  __builtin_amdgcn_s_wait_asynccnt(0);
#else
  asm volatile("s_wait_asynccnt 0" ::: "memory");
#endif
}
#endif

// ---------------- WMMA bf16 GEMM: OUT[M,N] = A[M,K] @ W[N,K]^T + bias ----------------
// A row-major [M,K] bf16, W row-major [N,K] bf16 (x @ W^T).
// Fragment layout (CDNA5 ISA 7.12.2, 16-bit A 16x32): lane l holds row (l&15),
// K runs [kh..kh+7] and [16+kh..16+kh+7], kh = (l>>4)*8. B mirrors with
// lane = output column. Each run is one contiguous 16-byte chunk.
//
// Block tile 128(M) x 128(N), 8 waves, each wave owns 32x64 (2x4 fragments).
// A/B K-slabs (128x32) staged through LDS with 16B row padding (stride 80B)
// so 16-lane fragment reads are bank-conflict free (20*r mod 64 full cycle).
// Staging uses GLOBAL_LOAD_ASYNC_TO_LDS_B128 (ASYNCcnt) when the toolchain
// exposes it, else global_load_b128 + ds_store_b128.
__device__ inline v16bf load_frag(const bf16_t* lo) {
  union U { v16bf v; v8bf h[2]; } u;
  u.h[0] = *(const v8bf*)(lo);
  u.h[1] = *(const v8bf*)(lo + 16);
  return u.v;
}

// EPI: 0 = f32 out (+bias), 1 = bf16 out (+bias, ReLU)
template <int EPI>
__global__ __launch_bounds__(256) void gemm_bf16(
    const bf16_t* __restrict__ A, const bf16_t* __restrict__ W,
    const float* __restrict__ bias, float* __restrict__ outf,
    bf16_t* __restrict__ outb, int Mdim, int Ndim, int K) {
  constexpr int LR = 40;                      // LDS row stride in elems (80 B)
  __shared__ bf16_t As[128 * LR];
  __shared__ bf16_t Bs[128 * LR];

  const int tid  = threadIdx.x;
  const int lane = tid & 31;
  const int wave = tid >> 5;                  // 0..7
  const int wm = wave & 3;                    // 4 waves along M
  const int wn = wave >> 2;                   // 2 waves along N
  const int blockM = blockIdx.x * 128;
  const int blockN = blockIdx.y * 128;
  const int rl = lane & 15;
  const int kh = (lane >> 4) * 8;

  // staging: 128 rows x 4 chunks (8 bf16 = 16B each) = 512 chunks, 2/thread/tile
  const int r0 = tid >> 2;                    // 0..63
  const int cc0 = (tid & 3) * 8;

  v8f c[2][4] = {};

  for (int k0 = 0; k0 < K; k0 += 32) {
    __syncthreads();
#if HAVE_ASYNC_LDS
#pragma unroll
    for (int s = 0; s < 2; ++s) {
      const int r = r0 + s * 64;
      __builtin_amdgcn_global_load_async_to_lds_b128(
          to_g128(A + (size_t)(blockM + r) * K + k0 + cc0),
          to_lds128(As + r * LR + cc0), 0, 0);
      __builtin_amdgcn_global_load_async_to_lds_b128(
          to_g128(W + (size_t)(blockN + r) * K + k0 + cc0),
          to_lds128(Bs + r * LR + cc0), 0, 0);
    }
#else
#pragma unroll
    for (int s = 0; s < 2; ++s) {
      const int r = r0 + s * 64;
      v8bf va = *(const v8bf*)(A + (size_t)(blockM + r) * K + k0 + cc0);
      v8bf vb = *(const v8bf*)(W + (size_t)(blockN + r) * K + k0 + cc0);
      *(v8bf*)(As + r * LR + cc0) = va;
      *(v8bf*)(Bs + r * LR + cc0) = vb;
    }
#endif
    if (k0 + 32 < K) {                        // gfx1250 global_prefetch path
      const int pr = tid >> 1;                // 0..127, every row covered
      __builtin_prefetch((const void*)(A + (size_t)(blockM + pr) * K + k0 + 32), 0, 1);
      __builtin_prefetch((const void*)(W + (size_t)(blockN + pr) * K + k0 + 32), 0, 1);
    }
#if HAVE_ASYNC_LDS
    wait_asynccnt0();
#endif
    __syncthreads();

    v16bf a[2], b[4];
#pragma unroll
    for (int fm = 0; fm < 2; ++fm)
      a[fm] = load_frag(As + (wm * 32 + fm * 16 + rl) * LR + kh);
#pragma unroll
    for (int fn = 0; fn < 4; ++fn)
      b[fn] = load_frag(Bs + (wn * 64 + fn * 16 + rl) * LR + kh);
#pragma unroll
    for (int fm = 0; fm < 2; ++fm)
#pragma unroll
      for (int fn = 0; fn < 4; ++fn)
        c[fm][fn] = __builtin_amdgcn_wmma_f32_16x16x32_bf16(
            false, a[fm], false, b[fn], (short)0, c[fm][fn], false, false);
  }

  const int half = lane >> 4;
#pragma unroll
  for (int fm = 0; fm < 2; ++fm)
#pragma unroll
    for (int fn = 0; fn < 4; ++fn) {
      const int n = blockN + wn * 64 + fn * 16 + rl;
      const float bv = bias[n];
#pragma unroll
      for (int r = 0; r < 8; ++r) {
        const int m = blockM + wm * 32 + fm * 16 + half * 8 + r;
        float val = c[fm][fn][r] + bv;
        if (EPI == 1) {
          val = fmaxf(val, 0.0f);
          outb[(size_t)m * Ndim + n] = (bf16_t)val;
        } else {
          outf[(size_t)m * Ndim + n] = val;
        }
      }
    }
}

// ---------------- elementwise / reduction kernels ----------------
__global__ void k_cvt(const float* __restrict__ in, bf16_t* __restrict__ out, int n) {
  int i = blockIdx.x * 256 + threadIdx.x;
  if (i < n) out[i] = (bf16_t)in[i];
}

__global__ void k_init(const float* __restrict__ data, const float* __restrict__ pos,
                       float* __restrict__ nodes, bf16_t* __restrict__ nodesb) {
  long long i = (long long)blockIdx.x * 256 + threadIdx.x;
  if (i >= kM * kH) return;
  int h = (int)(i % kH);
  int l = (int)((i / kH) % kL);
  float v = data[i] + pos[(size_t)l * kH + h];
  nodes[i] = v;
  nodesb[i] = (bf16_t)v;
}

__global__ void k_relay_init(const float* __restrict__ nodes, float* __restrict__ relay) {
  int b = blockIdx.x;
  int h = threadIdx.x;
  float s = 0.f;
  for (int l = 0; l < kL; ++l) s += nodes[((size_t)b * kL + l) * kH + h];
  relay[b * kH + h] = s / (float)kL;
}

__device__ inline float block_reduce_sum(float v, float* red) {
  int t = threadIdx.x;
  red[t] = v;
  __syncthreads();
  for (int s = 128; s > 0; s >>= 1) {
    if (t < s) red[t] += red[t + s];
    __syncthreads();
  }
  float r = red[0];
  __syncthreads();
  return r;
}

__device__ inline float block_reduce_max(float v, float* red) {
  int t = threadIdx.x;
  red[t] = v;
  __syncthreads();
  for (int s = 128; s > 0; s >>= 1) {
    if (t < s) red[t] = fmaxf(red[t], red[t + s]);
    __syncthreads();
  }
  float r = red[0];
  __syncthreads();
  return r;
}

// h = LN(t2 + res; g, b) -> outf (f32) + outb (bf16). One block per token row.
__global__ __launch_bounds__(256) void k_add_ln(
    const float* __restrict__ t2, const float* __restrict__ res,
    const float* __restrict__ g, const float* __restrict__ bb,
    float* __restrict__ outf, bf16_t* __restrict__ outb) {
  __shared__ float red[256];
  const size_t off = (size_t)blockIdx.x * kH;
  const int t = threadIdx.x;
  float x = t2[off + t] + res[off + t];
  float mean = block_reduce_sum(x, red) * (1.0f / kH);
  float d = x - mean;
  float var = block_reduce_sum(d * d, red) * (1.0f / kH);
  float y = d * rsqrtf(var + kEps) * g[t] + bb[t];
  outf[off + t] = y;
  outb[off + t] = (bf16_t)y;
}

// nodes = leaky_relu(LN(x; g, b)) -> f32 + bf16
__global__ __launch_bounds__(256) void k_ln_leaky(
    const float* __restrict__ xin, const float* __restrict__ g,
    const float* __restrict__ bb, float* __restrict__ outf,
    bf16_t* __restrict__ outb) {
  __shared__ float red[256];
  const size_t off = (size_t)blockIdx.x * kH;
  const int t = threadIdx.x;
  float x = xin[off + t];
  float mean = block_reduce_sum(x, red) * (1.0f / kH);
  float d = x - mean;
  float var = block_reduce_sum(d * d, red) * (1.0f / kH);
  float y = d * rsqrtf(var + kEps) * g[t] + bb[t];
  y = (y > 0.f) ? y : kSlope * y;
  outf[off + t] = y;
  outb[off + t] = (bf16_t)y;
}

// small dense: out[b,n] = act(sum_k in[b,k]*W[n,k] + bias[n]); ACT 0=none 1=leaky
template <int ACT>
__global__ void k_lin_small(const float* __restrict__ in, const float* __restrict__ W,
                            const float* __restrict__ bias, float* __restrict__ out,
                            int Bv, int N, int K) {
  int i = blockIdx.x * 256 + threadIdx.x;
  if (i >= Bv * N) return;
  int b = i / N, n = i % N;
  const float* ip = in + (size_t)b * K;
  const float* wp = W + (size_t)n * K;
  float s = bias[n];
  for (int k = 0; k < K; ++k) s += ip[k] * wp[k];
  if (ACT == 1) s = (s > 0.f) ? s : kSlope * s;
  out[i] = s;
}

// ring attention: window {l-1,l,l+1} (zero-padded) + relay key, softmax over 4
__global__ void k_ring_attn(const float* __restrict__ q, const float* __restrict__ k,
                            const float* __restrict__ v, const float* __restrict__ ak,
                            const float* __restrict__ av, bf16_t* __restrict__ att) {
  long long i = (long long)blockIdx.x * 256 + threadIdx.x;  // over B*L*NH
  if (i >= (long long)kB * kL * kNH) return;
  int n = (int)(i % kNH);
  int l = (int)((i / kNH) % kL);
  int b = (int)(i / ((long long)kNH * kL));
  const size_t row = ((size_t)b * kL + l) * kC + (size_t)n * kHD;
  const float* qp = q + row;
  float sc[4];
#pragma unroll
  for (int j = 0; j < 3; ++j) {
    int ll = l - 1 + j;
    float s = 0.f;
    if (ll >= 0 && ll < kL) {
      const float* kp = k + ((size_t)b * kL + ll) * kC + (size_t)n * kHD;
      for (int d = 0; d < kHD; ++d) s += qp[d] * kp[d];
    }
    sc[j] = s * kScale;  // zero-padded k => score 0 (still in softmax)
  }
  {
    const float* kp = ak + (size_t)b * kC + (size_t)n * kHD;
    float s = 0.f;
    for (int d = 0; d < kHD; ++d) s += qp[d] * kp[d];
    sc[3] = s * kScale;
  }
  float mx = fmaxf(fmaxf(sc[0], sc[1]), fmaxf(sc[2], sc[3]));
  float e0 = expf(sc[0] - mx), e1 = expf(sc[1] - mx);
  float e2 = expf(sc[2] - mx), e3 = expf(sc[3] - mx);
  float inv = 1.0f / (e0 + e1 + e2 + e3);
  const float* vm = (l - 1 >= 0) ? v + ((size_t)b * kL + (l - 1)) * kC + (size_t)n * kHD : nullptr;
  const float* vc = v + ((size_t)b * kL + l) * kC + (size_t)n * kHD;
  const float* vp = (l + 1 < kL) ? v + ((size_t)b * kL + (l + 1)) * kC + (size_t)n * kHD : nullptr;
  const float* vr = av + (size_t)b * kC + (size_t)n * kHD;
  for (int d = 0; d < kHD; ++d) {
    float o = e1 * vc[d] + e3 * vr[d];
    if (vm) o += e0 * vm[d];
    if (vp) o += e2 * vp[d];
    att[row + d] = (bf16_t)(o * inv);
  }
}

// star scores: s[b,n,l] over L+1 keys (key 0 = relay's own key)
__global__ void k_star_scores(const float* __restrict__ sq, const float* __restrict__ skr,
                              const float* __restrict__ skn,
                              const unsigned char* __restrict__ mask,
                              float* __restrict__ sbuf) {
  int bn = blockIdx.x;
  int b = bn / kNH, n = bn % kNH;
  const float* qv = sq + (size_t)b * kC + (size_t)n * kHD;
  for (int l = threadIdx.x; l < kL + 1; l += 256) {
    const float* kv = (l == 0) ? skr + (size_t)b * kC + (size_t)n * kHD
                               : skn + ((size_t)b * kL + (l - 1)) * kC + (size_t)n * kHD;
    float s = 0.f;
    for (int d = 0; d < kHD; ++d) s += qv[d] * kv[d];
    s *= kScale;
    if (l > 0 && !mask[b * kL + (l - 1)]) s = -INFINITY;
    sbuf[(size_t)bn * (kL + 1) + l] = s;
  }
}

// star softmax + weighted value sum -> satt[b, n*HD + d]
__global__ __launch_bounds__(256) void k_star_soft(
    const float* __restrict__ sbuf, const float* __restrict__ svr,
    const float* __restrict__ svn, float* __restrict__ satt) {
  __shared__ float red[256];
  int bn = blockIdx.x;
  int b = bn / kNH, n = bn % kNH;
  const float* s = sbuf + (size_t)bn * (kL + 1);
  const int t = threadIdx.x;
  float mx = -INFINITY;
  for (int l = t; l < kL + 1; l += 256) mx = fmaxf(mx, s[l]);
  mx = block_reduce_max(mx, red);
  float se = 0.f;
  for (int l = t; l < kL + 1; l += 256) se += expf(s[l] - mx);
  float denom = block_reduce_sum(se, red);
  const int d = t % kHD;      // 0..63
  const int grp = t / kHD;    // 0..3
  float acc = 0.f;
  for (int l = grp; l < kL + 1; l += 4) {
    float a = expf(s[l] - mx);
    const float* vv = (l == 0) ? svr + (size_t)b * kC + (size_t)n * kHD
                               : svn + ((size_t)b * kL + (l - 1)) * kC + (size_t)n * kHD;
    acc += a * vv[d];
  }
  red[t] = acc;
  __syncthreads();
  if (grp == 0) {
    float tot = red[d] + red[64 + d] + red[128 + d] + red[192 + d];
    satt[(size_t)b * kC + (size_t)n * kHD + d] = tot / denom;
  }
}

__global__ void k_pad_zero(const unsigned char* __restrict__ mask,
                           float* __restrict__ nodes, bf16_t* __restrict__ nodesb) {
  long long i = (long long)blockIdx.x * 256 + threadIdx.x;
  if (i >= kM * kH) return;
  long long bl = i / kH;
  if (!mask[bl]) {
    nodes[i] = 0.f;
    nodesb[i] = (bf16_t)0.f;
  }
}

__global__ void k_output(const float* __restrict__ nodes, const float* __restrict__ relay,
                         float* __restrict__ out) {
  long long i = (long long)blockIdx.x * 256 + threadIdx.x;
  const long long nn = kM * kH;
  if (i < nn) out[i] = nodes[i];
  else if (i < nn + (long long)kB * kH) out[i] = relay[i - nn];
}

// ---------------- host ----------------
extern "C" void kernel_launch(void* const* d_in, const int* in_sizes, int n_in,
                              void* d_out, int out_size, void* d_ws, size_t ws_size,
                              hipStream_t stream) {
  (void)in_sizes; (void)n_in; (void)out_size; (void)ws_size;
  const float* data          = (const float*)d_in[0];
  const unsigned char* mask  = (const unsigned char*)d_in[1];
  const float* pos_emb       = (const float*)d_in[2];
  const float* ln_g          = (const float*)d_in[3];
  const float* ln_b          = (const float*)d_in[4];
  const float* ff_w1         = (const float*)d_in[5];
  const float* ff_b1         = (const float*)d_in[6];
  const float* ff_w2         = (const float*)d_in[7];
  const float* ff_b2         = (const float*)d_in[8];
  const float* ff_lng        = (const float*)d_in[9];
  const float* ff_lnb        = (const float*)d_in[10];
  const float* ring_wq       = (const float*)d_in[11];
  const float* ring_bq       = (const float*)d_in[12];
  const float* ring_wk       = (const float*)d_in[13];
  const float* ring_bk       = (const float*)d_in[14];
  const float* ring_wv       = (const float*)d_in[15];
  const float* ring_bv       = (const float*)d_in[16];
  const float* ring_wo       = (const float*)d_in[17];
  const float* ring_bo       = (const float*)d_in[18];
  const float* star_wq       = (const float*)d_in[19];
  const float* star_bq       = (const float*)d_in[20];
  const float* star_wk       = (const float*)d_in[21];
  const float* star_bk       = (const float*)d_in[22];
  const float* star_wv       = (const float*)d_in[23];
  const float* star_bv       = (const float*)d_in[24];
  const float* star_wo       = (const float*)d_in[25];
  const float* star_bo       = (const float*)d_in[26];

  char* wsb = (char*)d_ws;
  size_t off = 0;
  auto take = [&](size_t bytes) -> char* {
    char* p = wsb + off;
    off = (off + bytes + 255) & ~(size_t)255;
    return p;
  };
  float*  nodes_f = (float*)take((size_t)kM * kH * 4);
  float*  buf1    = (float*)take((size_t)kM * kH * 4);   // t2 / ring_out
  float*  q_f     = (float*)take((size_t)kM * kC * 4);   // q, later star-sk(nodes)
  float*  k_f     = (float*)take((size_t)kM * kC * 4);   // k, later star-sv(nodes)
  float*  v_f     = (float*)take((size_t)kM * kC * 4);
  bf16_t* actbf   = (bf16_t*)take((size_t)kM * kC * 2);  // nodes_bf -> h_bf -> att_bf -> nodes_bf
  bf16_t* t1bf    = (bf16_t*)take((size_t)kM * kDI * 2);
  bf16_t* wbf     = (bf16_t*)take((size_t)kDI * kH * 2);
  float*  sbuf    = (float*)take((size_t)kB * kNH * (kL + 1) * 4);
  float*  relay_f = (float*)take((size_t)kB * kH * 4);
  float*  akbuf   = (float*)take((size_t)kB * kC * 4);
  float*  avbuf   = (float*)take((size_t)kB * kC * 4);
  float*  sqbuf   = (float*)take((size_t)kB * kC * 4);
  float*  skr     = (float*)take((size_t)kB * kC * 4);
  float*  svr     = (float*)take((size_t)kB * kC * 4);
  float*  satt    = (float*)take((size_t)kB * kC * 4);

  const long long NE = kM * kH;
  const int eltBlocks = (int)((NE + 255) / 256);

  k_init<<<eltBlocks, 256, 0, stream>>>(data, pos_emb, nodes_f, actbf);
  k_relay_init<<<kB, 256, 0, stream>>>(nodes_f, relay_f);

  const dim3 gFF1(kM / 128, kDI / 128);   // 256 x 8
  const dim3 gH(kM / 128, kH / 128);      // 256 x 2

  for (int i = 0; i < kNL; ++i) {
    const float* W1  = ff_w1 + (size_t)i * kDI * kH;
    const float* B1  = ff_b1 + (size_t)i * kDI;
    const float* W2  = ff_w2 + (size_t)i * kH * kDI;
    const float* B2  = ff_b2 + (size_t)i * kH;
    const float* LNG = ff_lng + (size_t)i * kH;
    const float* LNB = ff_lnb + (size_t)i * kH;
    const float* RWQ = ring_wq + (size_t)i * kC * kH; const float* RBQ = ring_bq + (size_t)i * kC;
    const float* RWK = ring_wk + (size_t)i * kC * kH; const float* RBK = ring_bk + (size_t)i * kC;
    const float* RWV = ring_wv + (size_t)i * kC * kH; const float* RBV = ring_bv + (size_t)i * kC;
    const float* RWO = ring_wo + (size_t)i * kH * kC; const float* RBO = ring_bo + (size_t)i * kH;
    const float* LG  = ln_g + (size_t)i * kH;
    const float* LB  = ln_b + (size_t)i * kH;
    const float* SWQ = star_wq + (size_t)i * kC * kH; const float* SBQ = star_bq + (size_t)i * kC;
    const float* SWK = star_wk + (size_t)i * kC * kH; const float* SBK = star_bk + (size_t)i * kC;
    const float* SWV = star_wv + (size_t)i * kC * kH; const float* SBV = star_bv + (size_t)i * kC;
    const float* SWO = star_wo + (size_t)i * kH * kC; const float* SBO = star_bo + (size_t)i * kH;

    // ---- FFN: t1 = relu(nodes @ W1^T + b1);  t2 = t1 @ W2^T + b2;  h = LN(t2 + nodes) ----
    k_cvt<<<(kDI * kH + 255) / 256, 256, 0, stream>>>(W1, wbf, kDI * kH);
    gemm_bf16<1><<<gFF1, 256, 0, stream>>>(actbf, wbf, B1, nullptr, t1bf, (int)kM, kDI, kH);
    k_cvt<<<(kH * kDI + 255) / 256, 256, 0, stream>>>(W2, wbf, kH * kDI);
    gemm_bf16<0><<<gH, 256, 0, stream>>>(t1bf, wbf, B2, buf1, nullptr, (int)kM, kH, kDI);
    k_add_ln<<<(int)kM, 256, 0, stream>>>(buf1, nodes_f, LNG, LNB, buf1, actbf);

    // ---- ring attention ----
    k_cvt<<<(kC * kH + 255) / 256, 256, 0, stream>>>(RWQ, wbf, kC * kH);
    gemm_bf16<0><<<gH, 256, 0, stream>>>(actbf, wbf, RBQ, q_f, nullptr, (int)kM, kC, kH);
    k_cvt<<<(kC * kH + 255) / 256, 256, 0, stream>>>(RWK, wbf, kC * kH);
    gemm_bf16<0><<<gH, 256, 0, stream>>>(actbf, wbf, RBK, k_f, nullptr, (int)kM, kC, kH);
    k_cvt<<<(kC * kH + 255) / 256, 256, 0, stream>>>(RWV, wbf, kC * kH);
    gemm_bf16<0><<<gH, 256, 0, stream>>>(actbf, wbf, RBV, v_f, nullptr, (int)kM, kC, kH);
    k_lin_small<0><<<(kB * kC + 255) / 256, 256, 0, stream>>>(relay_f, RWK, RBK, akbuf, kB, kC, kH);
    k_lin_small<0><<<(kB * kC + 255) / 256, 256, 0, stream>>>(relay_f, RWV, RBV, avbuf, kB, kC, kH);
    k_ring_attn<<<(int)(((long long)kB * kL * kNH + 255) / 256), 256, 0, stream>>>(
        q_f, k_f, v_f, akbuf, avbuf, actbf);
    k_cvt<<<(kH * kC + 255) / 256, 256, 0, stream>>>(RWO, wbf, kH * kC);
    gemm_bf16<0><<<gH, 256, 0, stream>>>(actbf, wbf, RBO, buf1, nullptr, (int)kM, kH, kC);
    k_ln_leaky<<<(int)kM, 256, 0, stream>>>(buf1, LG, LB, nodes_f, actbf);

    // ---- star attention (relay attends to [relay; nodes]) ----
    k_lin_small<0><<<(kB * kC + 255) / 256, 256, 0, stream>>>(relay_f, SWQ, SBQ, sqbuf, kB, kC, kH);
    k_lin_small<0><<<(kB * kC + 255) / 256, 256, 0, stream>>>(relay_f, SWK, SBK, skr, kB, kC, kH);
    k_lin_small<0><<<(kB * kC + 255) / 256, 256, 0, stream>>>(relay_f, SWV, SBV, svr, kB, kC, kH);
    k_cvt<<<(kC * kH + 255) / 256, 256, 0, stream>>>(SWK, wbf, kC * kH);
    gemm_bf16<0><<<gH, 256, 0, stream>>>(actbf, wbf, SBK, q_f, nullptr, (int)kM, kC, kH);
    k_cvt<<<(kC * kH + 255) / 256, 256, 0, stream>>>(SWV, wbf, kC * kH);
    gemm_bf16<0><<<gH, 256, 0, stream>>>(actbf, wbf, SBV, k_f, nullptr, (int)kM, kC, kH);
    k_star_scores<<<kB * kNH, 256, 0, stream>>>(sqbuf, skr, q_f, mask, sbuf);
    k_star_soft<<<kB * kNH, 256, 0, stream>>>(sbuf, svr, k_f, satt);
    k_lin_small<1><<<(kB * kH + 255) / 256, 256, 0, stream>>>(satt, SWO, SBO, relay_f, kB, kH, kC);

    // ---- pad zeroing (after star, matching reference order) ----
    k_pad_zero<<<eltBlocks, 256, 0, stream>>>(mask, nodes_f, actbf);
  }

  const long long totalOut = NE + (long long)kB * kH;
  k_output<<<(int)((totalOut + 255) / 256), 256, 0, stream>>>(nodes_f, relay_f, (float*)d_out);
}
